// DFConv2d_20220706029732
// MI455X (gfx1250) — matrix operand order repaired and checked by reading, not compiled, not run on hardware
//
#include <hip/hip_runtime.h>
#include <math.h>

// ---- problem constants ----
#define BB    4
#define CIN   256
#define HH    64
#define WW    64
#define COUT  256
#define HWSZ  (HH*WW)            // 4096
#define NPIX  (BB*HWSZ)          // 16384
#define KRED  (9*CIN)            // 2304
#define KSTEPS (KRED/32)         // 72

typedef __attribute__((ext_vector_type(16))) __bf16 v16bf;
typedef __attribute__((ext_vector_type(8)))  __bf16 v8bf;
typedef __attribute__((ext_vector_type(4)))  __bf16 v4bf;
typedef __attribute__((ext_vector_type(8)))  float  v8f;

// ---------------------------------------------------------------------------
// Kernel T: NCHW fp32 -> NHWC bf16 transpose through LDS (coalesced both ways).
// 32ch x 32pix tile per 256-thread block.
// ---------------------------------------------------------------------------
__global__ __launch_bounds__(256) void dfc_transpose(const float* __restrict__ x,
                                                     __bf16* __restrict__ xT)
{
    __shared__ __bf16 tile[32][34];
    int blk    = blockIdx.x;                 // 8 c-tiles x 512 pix-tiles = 4096
    int ctile  = blk & 7;
    int pixt   = blk >> 3;
    int c0     = ctile * 32;
    int pix0   = pixt * 32;                  // global pixel (b*4096 + h*64 + w)
    int t      = threadIdx.x;
    int ht     = t & 31;
    int cq     = t >> 5;                     // 0..7
    int b      = pix0 >> 12;
    int hwl    = pix0 & 4095;
    const float* xsrc = x + (size_t)b * CIN * HWSZ;
#pragma unroll
    for (int i = 0; i < 4; ++i) {
        int ct = cq * 4 + i;
        tile[ct][ht] = (__bf16)xsrc[(size_t)(c0 + ct) * HWSZ + hwl + ht];
    }
    __syncthreads();
    int cw = t & 31;
#pragma unroll
    for (int i = 0; i < 4; ++i) {
        int pt = cq * 4 + i;
        xT[(size_t)(pix0 + pt) * CIN + c0 + cw] = tile[cw][pt];
    }
}

// ---------------------------------------------------------------------------
// Kernel P: pack an OIHW fp32 weight (nrows x 256 x 3 x 3) into bf16 WMMA
// A-fragments, rows >= nrows zero-padded.
// Layout: dst[mtile][kstep][lane][j], j = v16bf element index.
//   lanes 0-15 : M=lane,    K=kbase+j
//   lanes 16-31: M=lane-16, K=kbase+16+j
// K index = tap*256 + c (tap = kh*3+kw), matching B staging below.
// ---------------------------------------------------------------------------
__global__ __launch_bounds__(256) void dfc_pack(const float* __restrict__ src,
                                                __bf16* __restrict__ dst,
                                                int total, int nrows)
{
    int idx = blockIdx.x * 256 + threadIdx.x;
    if (idx >= total) return;
    int j     = idx & 15;
    int lane  = (idx >> 4) & 31;
    int kstep = (idx >> 9) % KSTEPS;
    int mtile = idx / (KSTEPS * 32 * 16);
    int m     = lane & 15;
    bool lo   = lane < 16;
    int kbase = kstep * 32;
    int k;
    k = kbase + (lo ? j : j + 16);
    int tap = k >> 8;
    int c   = k & 255;
    int oc  = mtile * 16 + m;
    float v = (oc < nrows) ? src[(oc * CIN + c) * 9 + tap] : 0.f;
    dst[idx] = (__bf16)v;
}

// ---------------------------------------------------------------------------
// Kernel 1: offset/mask conv as a WMMA GEMM. M=32 (27 valid), K=2304, N=16.
// 64 threads (2 waves, one M-tile each). B tile staged from NHWC-bf16 x with
// plain zero-padded taps (single b128 copy per thread per kstep).
// Epilogue goes through LDS to compute py/px/sigmoid(mask) per pixel.
// ---------------------------------------------------------------------------
__global__ __launch_bounds__(64) void dfc_offsets_wmma(const __bf16* __restrict__ xT,
                                                       const __bf16* __restrict__ ApkOff,
                                                       const float* __restrict__ b_off,
                                                       float* __restrict__ PY,
                                                       float* __restrict__ PX,
                                                       float* __restrict__ MK)
{
    __shared__ __align__(32) __bf16 ldsB[2][512];
    __shared__ float om[32 * 16];

    const int t      = threadIdx.x;
    const int lane   = t & 31;
    const int waveId = t >> 5;                 // 0..1 -> oc 0-15 / 16-31
    const int p0     = blockIdx.x * 16;
    const int n      = t & 15;
    const int g      = t >> 4;                 // 0..3, 8 channels each
    const int p      = p0 + n;
    const int b      = p >> 12;
    const int h      = (p >> 6) & 63;
    const int w      = p & 63;

    // B-fragment slot: c<16 -> lane=n, e=c ; c>=16 -> lane=n+16, e=c-16
    const int wroff = ((g >= 2) ? (n + 16) : n) * 16 + (g & 1) * 8;

    v8bf zero8;
#pragma unroll
    for (int i = 0; i < 8; ++i) zero8[i] = (__bf16)0.f;

    v8f acc = (v8f){0.f,0.f,0.f,0.f,0.f,0.f,0.f,0.f};

    for (int tap = 0; tap < 9; ++tap) {
        int kh = tap / 3, kw = tap % 3;
        int y  = h + kh - 1;
        int xx = w + kw - 1;
        bool valid = (y >= 0) & (y < HH) & (xx >= 0) & (xx < WW);
        int pix = b * HWSZ + (valid ? (y * WW + xx) : 0);
        const v8bf* src = (const v8bf*)(xT + (size_t)pix * CIN);
        for (int cblk = 0; cblk < 8; ++cblk) {
            int ks = tap * 8 + cblk;
            v8bf v = valid ? src[cblk * 4 + g] : zero8;
            *(v8bf*)&ldsB[ks & 1][wroff] = v;
            __syncthreads();
            v16bf bfrag = *(const v16bf*)&ldsB[ks & 1][lane * 16];
            const __bf16* ap =
                ApkOff + (((size_t)waveId * KSTEPS + ks) * 32 + lane) * 16;
            v16bf af = *(const v16bf*)ap;
            acc = __builtin_amdgcn_wmma_f32_16x16x32_bf16(
                false, af, false, bfrag, (short)0, acc, false, false);
        }
    }

    // D layout: VGPR v -> M=v (lanes 0-15) / M=8+v (lanes 16-31), N=lane%16
#pragma unroll
    for (int v = 0; v < 8; ++v) {
        int m = v + ((lane >= 16) ? 8 : 0);
        om[(waveId * 16 + m) * 16 + (lane & 15)] = acc[v];
    }
    __syncthreads();

    if (t < 16) {
        int p2 = p0 + t;
        int h2 = (p2 >> 6) & 63, w2 = p2 & 63;
#pragma unroll
        for (int tap = 0; tap < 9; ++tap) {
            float dy = om[(2 * tap) * 16 + t]     + b_off[2 * tap];
            float dx = om[(2 * tap + 1) * 16 + t] + b_off[2 * tap + 1];
            float mm = om[(18 + tap) * 16 + t]    + b_off[18 + tap];
            mm = 1.f / (1.f + expf(-mm));
            PY[tap * NPIX + p2] = (float)(h2 - 1 + tap / 3) + dy;
            PX[tap * NPIX + p2] = (float)(w2 - 1 + tap % 3) + dx;
            MK[tap * NPIX + p2] = mm;
        }
    }
}

// ---------------------------------------------------------------------------
// Kernel 2: fused bilinear sample + modulate + GEMM (bf16 WMMA, f32 acc).
// 256 threads (8 waves). WG = 32 pixels (2 N-tiles) x 256 COUT; each wave
// owns 2 M-tiles x 2 N-tiles (4 WMMAs / kstep, 32 acc VGPRs). Doubling N per
// WG halves the packed-A L2 re-fetch (~0.6 GB total) vs a 16-pixel tile.
// Corner gathers are 8B (4 consecutive bf16 channels) from NHWC x; double-
// buffered LDS B tiles -> one barrier per kstep.
// ---------------------------------------------------------------------------
__global__ __launch_bounds__(256) void dfc_main(const __bf16* __restrict__ xT,
                                                const float* __restrict__ PY,
                                                const float* __restrict__ PX,
                                                const float* __restrict__ MK,
                                                const __bf16* __restrict__ Apk,
                                                float* __restrict__ out)
{
    __shared__ __align__(32) __bf16 ldsB[2][2][512];   // [buf][ntile][frag]

    const int t      = threadIdx.x;
    const int lane   = t & 31;
    const int waveId = t >> 5;                 // 0..7, owns mtiles 2w,2w+1
    const int p0     = blockIdx.x * 32;
    const int n      = t & 31;                 // pixel within WG tile
    const int g      = t >> 5;                 // 0..7, 4 channels each
    const int p      = p0 + n;
    const int b      = p >> 12;

    // B-fragment slot for staging: ntile = n>>4, within-tile pixel nn = n&15
    const int wroff = (n >> 4) * 512 + ((n & 15) + ((g >= 4) ? 16 : 0)) * 16
                    + (g & 3) * 4;

    v8f acc[2][2];
#pragma unroll
    for (int i = 0; i < 2; ++i)
#pragma unroll
        for (int j = 0; j < 2; ++j)
            acc[i][j] = (v8f){0.f,0.f,0.f,0.f,0.f,0.f,0.f,0.f};

    for (int tap = 0; tap < 9; ++tap) {
        float py = PY[tap * NPIX + p];
        float px = PX[tap * NPIX + p];
        float mm = MK[tap * NPIX + p];
        float y0f = floorf(py), x0f = floorf(px);
        float wy = py - y0f, wx = px - x0f;
        int y0 = (int)y0f, x0 = (int)x0f;
        int y1 = y0 + 1,  x1 = x0 + 1;
        float vy0 = (y0 >= 0 && y0 < HH) ? 1.f : 0.f;
        float vy1 = (y1 >= 0 && y1 < HH) ? 1.f : 0.f;
        float vx0 = (x0 >= 0 && x0 < WW) ? 1.f : 0.f;
        float vx1 = (x1 >= 0 && x1 < WW) ? 1.f : 0.f;
        int yc0 = min(max(y0, 0), HH - 1), yc1 = min(max(y1, 0), HH - 1);
        int xc0 = min(max(x0, 0), WW - 1), xc1 = min(max(x1, 0), WW - 1);
        float w00 = (1.f - wy) * (1.f - wx) * mm * vy0 * vx0;
        float w01 = (1.f - wy) * wx         * mm * vy0 * vx1;
        float w10 = wy         * (1.f - wx) * mm * vy1 * vx0;
        float w11 = wy         * wx         * mm * vy1 * vx1;
        const __bf16* q00 = xT + (size_t)(b * HWSZ + yc0 * WW + xc0) * CIN;
        const __bf16* q01 = xT + (size_t)(b * HWSZ + yc0 * WW + xc1) * CIN;
        const __bf16* q10 = xT + (size_t)(b * HWSZ + yc1 * WW + xc0) * CIN;
        const __bf16* q11 = xT + (size_t)(b * HWSZ + yc1 * WW + xc1) * CIN;

        for (int cblk = 0; cblk < 8; ++cblk) {
            int ks = tap * 8 + cblk;
            int cb = cblk * 32 + (g << 2);
            // ---- stage: 4 channels, one 8B load per corner ----
            v4bf a00 = *(const v4bf*)(q00 + cb);
            v4bf a01 = *(const v4bf*)(q01 + cb);
            v4bf a10 = *(const v4bf*)(q10 + cb);
            v4bf a11 = *(const v4bf*)(q11 + cb);
            v4bf r;
#pragma unroll
            for (int i = 0; i < 4; ++i) {
                float v = w00 * (float)a00[i] + w01 * (float)a01[i]
                        + w10 * (float)a10[i] + w11 * (float)a11[i];
                r[i] = (__bf16)v;
            }
            *(v4bf*)&ldsB[ks & 1][0][wroff] = r;
            __syncthreads();

            // ---- consume: 2 M-tiles x 2 N-tiles per wave ----
            v16bf bf0 = *(const v16bf*)&ldsB[ks & 1][0][lane * 16];
            v16bf bf1 = *(const v16bf*)&ldsB[ks & 1][1][lane * 16];
            const __bf16* ap =
                Apk + (((size_t)(waveId * 2) * KSTEPS + ks) * 32 + lane) * 16;
            __builtin_prefetch((const void*)(ap + 512), 0, 1);  // next kstep A
#pragma unroll
            for (int mt = 0; mt < 2; ++mt) {
                v16bf af = *(const v16bf*)(ap + (size_t)mt * (KSTEPS * 32 * 16));
                acc[mt][0] = __builtin_amdgcn_wmma_f32_16x16x32_bf16(
                    false, af, false, bf0, (short)0, acc[mt][0], false, false);
                acc[mt][1] = __builtin_amdgcn_wmma_f32_16x16x32_bf16(
                    false, af, false, bf1, (short)0, acc[mt][1], false, false);
            }
            // double-buffered: next iteration writes the other LDS buffer
        }
    }

    // ---- epilogue: scatter D tiles to out (B,COUT,H,W) fp32 ----
#pragma unroll
    for (int nt = 0; nt < 2; ++nt) {
        int pp  = p0 + nt * 16 + (lane & 15);
        int rem = pp & 4095;
        float* ob = out + (size_t)b * COUT * HWSZ + rem;
#pragma unroll
        for (int mt = 0; mt < 2; ++mt) {
            int ocb = (waveId * 2 + mt) * 16 + ((lane >= 16) ? 8 : 0);
#pragma unroll
            for (int v = 0; v < 8; ++v)
                ob[(ocb + v) * HWSZ] = acc[mt][nt][v];
        }
    }
}

// ---------------------------------------------------------------------------
extern "C" void kernel_launch(void* const* d_in, const int* in_sizes, int n_in,
                              void* d_out, int out_size, void* d_ws, size_t ws_size,
                              hipStream_t stream)
{
    const float* x      = (const float*)d_in[0];       // (4,256,64,64)
    const float* w_off  = (const float*)d_in[1];       // (27,256,3,3)
    const float* b_off  = (const float*)d_in[2];       // (27,)
    const float* w_conv = (const float*)d_in[3];       // (256,256,3,3)
    float* out          = (float*)d_out;               // (4,256,64,64) fp32

    char* ws = (char*)d_ws;
    __bf16* xT     = (__bf16*)(ws);                    //  8,388,608 B NHWC bf16
    float*  PY     = (float*)(ws + 8388608);           //    589,824 B
    float*  PX     = (float*)(ws + 8978432);           //    589,824 B
    float*  MK     = (float*)(ws + 9568256);           //    589,824 B
    __bf16* Apk    = (__bf16*)(ws + 10158080);         //  1,179,648 B (32B aligned)
    __bf16* ApkOff = (__bf16*)(ws + 11337728);         //    147,456 B (32B aligned)

    // 1) x: NCHW fp32 -> NHWC bf16 (L2-resident, 8B/16B channel-vector gathers)
    dfc_transpose<<<4096, 256, 0, stream>>>(x, xT);
    // 2) pack both weight tensors into WMMA A-fragment layout (bf16)
    dfc_pack<<<(16 * KSTEPS * 32 * 16 + 255) / 256, 256, 0, stream>>>(
        w_conv, Apk, 16 * KSTEPS * 32 * 16, COUT);
    dfc_pack<<<(2 * KSTEPS * 32 * 16 + 255) / 256, 256, 0, stream>>>(
        w_off, ApkOff, 2 * KSTEPS * 32 * 16, 27);
    // 3) offset/mask conv via WMMA -> sampling coordinates + sigmoid mask
    dfc_offsets_wmma<<<NPIX / 16, 64, 0, stream>>>(xT, ApkOff, b_off, PY, PX, MK);
    // 4) fused bilinear sampling + modulation + bf16 WMMA GEMM
    dfc_main<<<NPIX / 32, 256, 0, stream>>>(xT, PY, PX, MK, Apk, out);
}
